// GGNN_2723009266381
// MI455X (gfx1250) — matrix operand order repaired and checked by reading, not compiled, not run on hardware
//
#include <hip/hip_runtime.h>
#include <hip/hip_bf16.h>

// Problem constants (from reference)
#define BB 16
#define NN 512
#define EE 4
#define HH 64
#define AD 32
#define STEPS_K 5
#define KSTEPS (EE * NN / 32)   // 64 k-steps of 32 in the aggregation GEMM

typedef __attribute__((ext_vector_type(16))) __bf16 bf16x16;
typedef __attribute__((ext_vector_type(8)))  __bf16 bf16x8;
typedef __attribute__((ext_vector_type(8)))  float  f32x8;
typedef __attribute__((ext_vector_type(4)))  float  f32x4;
typedef __bf16 bf16;

#define WMMA_BF16(A_, B_, C_) \
  __builtin_amdgcn_wmma_f32_16x16x32_bf16(false, (A_), false, (B_), (short)0, (C_), false, false)

// CDNA5 async global->LDS copy path (ASYNCcnt), if the toolchain exposes it.
#if defined(__has_builtin)
#if __has_builtin(__builtin_amdgcn_global_load_async_to_lds_b128)
#define HAVE_ASYNC_LDS 1
#endif
#endif

#ifdef HAVE_ASYNC_LDS
// Parameter types per hipcc diagnostic: (int4 __device__*, int4 __shared__*, imm, imm)
typedef int int4v __attribute__((__vector_size__(4 * sizeof(int))));
typedef __attribute__((address_space(1))) int4v* as1_int4p;
typedef __attribute__((address_space(3))) int4v* as3_int4p;
#define ASYNC_LD_B128(g_, l_)                                              \
  __builtin_amdgcn_global_load_async_to_lds_b128(                          \
      (as1_int4p)(void*)(g_), (as3_int4p)(l_), 0, 0)
#define WAIT_ASYNC() asm volatile("s_wait_asynccnt 0x0" ::: "memory")
#endif

static __device__ __forceinline__ bf16x16 cat8(bf16x8 a, bf16x8 b) {
  bf16x16 r;
#pragma unroll
  for (int i = 0; i < 8; ++i) { r[i] = a[i]; r[i + 8] = b[i]; }
  return r;
}

// A-operand fragment (16x32 tile) from a bf16 row-major matrix.
// Lanes 0-15 hold K = {k0..k0+7} U {k0+16..k0+23}; lanes 16-31 the other halves.
static __device__ __forceinline__ bf16x16 afrag_bf(const bf16* row, int k0, int hh) {
  const bf16* p = row + k0 + hh * 8;
  bf16x8 r0 = *(const bf16x8*)(p);
  bf16x8 r1 = *(const bf16x8*)(p + 16);
  return cat8(r0, r1);
}

// A-fragment sourced from f32 row-major data; converts to bf16 in registers.
static __device__ __forceinline__ bf16x16 afrag_f32(const float* row, int k0, int hh) {
  const float* p = row + k0 + hh * 8;
  f32x4 a = *(const f32x4*)(p);
  f32x4 b = *(const f32x4*)(p + 4);
  f32x4 c = *(const f32x4*)(p + 16);
  f32x4 d = *(const f32x4*)(p + 20);
  bf16x16 r;
#pragma unroll
  for (int i = 0; i < 4; ++i) {
    r[i]      = (bf16)a[i];
    r[i + 4]  = (bf16)b[i];
    r[i + 8]  = (bf16)c[i];
    r[i + 12] = (bf16)d[i];
  }
  return r;
}

// B-operand fragment (32x16 tile) from K-contiguous storage (col = column n's K-run).
static __device__ __forceinline__ bf16x16 bfrag(const bf16* col, int k0, int hh) {
  const bf16* p = col + k0 + hh * 8;
  bf16x8 r0 = *(const bf16x8*)(p);
  bf16x8 r1 = *(const bf16x8*)(p + 16);
  return cat8(r0, r1);
}

static __device__ __forceinline__ float sigf(float x) {
  return 1.0f / (1.0f + __expf(-x));
}

// ---------------- prep kernels (run once per launch) ----------------

__global__ void k_cvt_bf16(const float* __restrict__ src, bf16* __restrict__ dst, int total) {
  int i = blockIdx.x * blockDim.x + threadIdx.x;
  if (i < total) dst[i] = (bf16)src[i];
}

// dst[b][n][k] = src[b][k][n]  (bf16 K-contiguous weight layout)
__global__ void k_tr_w(const float* __restrict__ src, bf16* __restrict__ dst,
                       int batch, int K, int N) {
  int i = blockIdx.x * blockDim.x + threadIdx.x;
  int total = batch * K * N;
  if (i >= total) return;
  int k = i % K;
  int n = (i / K) % N;
  int b = i / (K * N);
  dst[i] = (bf16)src[((size_t)b * K + k) * N + n];
}

// dst[b][n][k] = src[b][k][n]  (f32 transposed hidden state: hsT[b][h][n])
__global__ void k_tr_f32(const float* __restrict__ src, float* __restrict__ dst,
                         int batch, int K, int N) {
  int i = blockIdx.x * blockDim.x + threadIdx.x;
  int total = batch * K * N;
  if (i >= total) return;
  int k = i % K;
  int n = (i / K) % N;
  int b = i / (K * N);
  dst[i] = src[((size_t)b * K + k) * N + n];
}

// ---------------- per-step kernels ----------------

// inc/outg: per (b,e): hs[b] (512x64) @ in_w[e] (64x64) + bias -> stored TRANSPOSED
// as incT[b][h][e*512+n] (bf16): K-contiguous B-operand for the big GEMM.
// 2048 waves: b(16) x e(4) x ntile(32). blockDim=256.
__global__ void k_inout(const bf16* __restrict__ hs_bf,
                        const bf16* __restrict__ w_inT, const bf16* __restrict__ w_outT,
                        const float* __restrict__ in_b, const float* __restrict__ out_b,
                        bf16* __restrict__ incT, bf16* __restrict__ outgT) {
  int wid  = (blockIdx.x * blockDim.x + threadIdx.x) >> 5;
  int lane = threadIdx.x & 31;
  int l15  = lane & 15;
  int hh   = lane >> 4;
  int ntile = wid & 31;
  int e     = (wid >> 5) & 3;
  int b     = wid >> 7;

  const bf16* arow = hs_bf + ((size_t)(b * NN + ntile * 16 + l15)) * HH;
  bf16x16 a0 = afrag_bf(arow, 0, hh);
  bf16x16 a1 = afrag_bf(arow, 32, hh);

#pragma unroll
  for (int pass = 0; pass < 2; ++pass) {
    const bf16*  wT   = (pass ? w_outT : w_inT) + (size_t)e * HH * HH;
    const float* bias = (pass ? out_b : in_b) + e * HH;
    bf16* dstT = (pass ? outgT : incT) + (size_t)b * HH * (EE * NN);
#pragma unroll
    for (int c = 0; c < 4; ++c) {
      int h = c * 16 + l15;
      float bv = bias[h];
      f32x8 acc;
#pragma unroll
      for (int j = 0; j < 8; ++j) acc[j] = bv;
      const bf16* col = wT + (size_t)h * HH;
      acc = WMMA_BF16(a0, bfrag(col, 0, hh), acc);
      acc = WMMA_BF16(a1, bfrag(col, 32, hh), acc);
      int m0 = e * NN + ntile * 16 + hh * 8;
      bf16x8 o;
#pragma unroll
      for (int j = 0; j < 8; ++j) o[j] = (bf16)acc[j];
      *(bf16x8*)(dstT + (size_t)h * (EE * NN) + m0) = o;
    }
  }
}

// Big aggregation GEMM: a_in/a_out[b] (512x64) = A_{in,out}[b] (512x2048) @ inc/outg.
// All 8 waves of a block share (b,dir) => the 64x32 B panel per k-step is staged
// once in LDS (double-buffered; async global->LDS copy when available).
// Rows padded to 40 bf16 (20 banks) for conflict-free ds_load_b128 fragment reads.
// 128 blocks x 256 threads: block = (b, dir, ntile-group-of-8).
__global__ void k_agg(const float* __restrict__ Amat,
                      const bf16* __restrict__ incT, const bf16* __restrict__ outgT,
                      bf16* __restrict__ ain_bf) {
  __shared__ bf16 bpanel[2][64 * 40];

  int tid  = threadIdx.x;
  int wave = tid >> 5;
  int lane = tid & 31;
  int l15  = lane & 15;
  int hh   = lane >> 4;
  int ntile = (blockIdx.x & 3) * 8 + wave;
  int dir   = (blockIdx.x >> 2) & 1;
  int b     = blockIdx.x >> 3;

  const float* arow = Amat + ((size_t)(b * NN + ntile * 16 + l15)) * (2 * EE * NN)
                           + dir * (EE * NN);
  const bf16* T = (dir ? outgT : incT) + (size_t)b * HH * (EE * NN);

  // Panel staging: thread tid covers column ph (0..63), 8-element segment pseg (0..3).
  int ph = tid >> 2, pseg = tid & 3;
  const bf16* psrc = T + (size_t)ph * (EE * NN) + pseg * 8;
  bf16* pdst0 = &bpanel[0][ph * 40 + pseg * 8];
  bf16* pdst1 = &bpanel[1][ph * 40 + pseg * 8];

  f32x8 acc[4];
#pragma unroll
  for (int c = 0; c < 4; ++c)
#pragma unroll
    for (int j = 0; j < 8; ++j) acc[c][j] = 0.0f;

#ifdef HAVE_ASYNC_LDS
  ASYNC_LD_B128(psrc, pdst0);
  for (int ks = 0; ks < KSTEPS; ++ks) {
    int k0 = ks * 32;
    WAIT_ASYNC();
    __syncthreads();
    if (ks + 1 < KSTEPS) ASYNC_LD_B128(psrc + k0 + 32, (ks & 1) ? pdst0 : pdst1);
    if (k0 + 288 <= EE * NN) __builtin_prefetch(arow + k0 + 256, 0, 0);
    bf16x16 af = afrag_f32(arow, k0, hh);
    const bf16* cur = &bpanel[ks & 1][0];
#pragma unroll
    for (int c = 0; c < 4; ++c)
      acc[c] = WMMA_BF16(af, bfrag(cur + (c * 16 + l15) * 40, 0, hh), acc[c]);
  }
#else
  {
    bf16x8 stage = *(const bf16x8*)psrc;
    *(bf16x8*)pdst0 = stage;
  }
  for (int ks = 0; ks < KSTEPS; ++ks) {
    int k0 = ks * 32;
    __syncthreads();
    bf16x8 nstage;
    if (ks + 1 < KSTEPS) nstage = *(const bf16x8*)(psrc + k0 + 32);
    if (k0 + 288 <= EE * NN) __builtin_prefetch(arow + k0 + 256, 0, 0);
    bf16x16 af = afrag_f32(arow, k0, hh);
    const bf16* cur = &bpanel[ks & 1][0];
#pragma unroll
    for (int c = 0; c < 4; ++c)
      acc[c] = WMMA_BF16(af, bfrag(cur + (c * 16 + l15) * 40, 0, hh), acc[c]);
    if (ks + 1 < KSTEPS) *(bf16x8*)((ks & 1) ? pdst0 : pdst1) = nstage;
  }
#endif

  // Store bf16 row-major into ain_bf (row stride 128): cols [0,64)=a_in, [64,128)=a_out.
#pragma unroll
  for (int c = 0; c < 4; ++c) {
    int h = dir * HH + c * 16 + l15;
#pragma unroll
    for (int j = 0; j < 8; ++j) {
      int n = ntile * 16 + hh * 8 + j;
      ain_bf[((size_t)(b * NN + n)) * (2 * HH) + h] = (bf16)acc[c][j];
    }
  }
}

// Fused GRU step: z = sig(a@Wz+b), r = sig(a@Wr+b), rh = r*hs (via LDS transpose),
// h' = tanh([a_in,a_out,rh]@Wt+b), hs = (1-z)*hs + z*h'. Updates hsT (f32) + hs_bf.
// 512 waves: b(16) x ntile(32). blockDim=256, LDS rows padded to 72 (36 banks).
__global__ void k_gru(const bf16* __restrict__ ain_bf, bf16* __restrict__ hs_bf,
                      const bf16* __restrict__ w_updT, const float* __restrict__ upd_b,
                      const bf16* __restrict__ w_rstT, const float* __restrict__ rst_b,
                      const bf16* __restrict__ w_trnT, const float* __restrict__ trn_b,
                      float* __restrict__ hsT) {
  __shared__ bf16 rhtile[8][16 * 72];

  int wave = threadIdx.x >> 5;
  int lane = threadIdx.x & 31;
  int l15  = lane & 15;
  int hh   = lane >> 4;
  int wid  = blockIdx.x * 8 + wave;
  int ntile = wid & 31;
  int b     = wid >> 5;

  const bf16* arow = ain_bf + ((size_t)(b * NN + ntile * 16 + l15)) * (2 * HH);
  const bf16* hrow = hs_bf + ((size_t)(b * NN + ntile * 16 + l15)) * HH;
  bf16x16 afr[6];
  afr[0] = afrag_bf(arow, 0, hh);
  afr[1] = afrag_bf(arow, 32, hh);
  afr[2] = afrag_bf(arow, 64, hh);
  afr[3] = afrag_bf(arow, 96, hh);
  afr[4] = afrag_bf(hrow, 0, hh);
  afr[5] = afrag_bf(hrow, 32, hh);

  // ---- z gate (kept in accumulator-layout registers) ----
  f32x8 zfr[4];
#pragma unroll
  for (int c = 0; c < 4; ++c) {
    int h = c * 16 + l15;
    float bv = upd_b[h];
    f32x8 acc;
#pragma unroll
    for (int j = 0; j < 8; ++j) acc[j] = bv;
    const bf16* col = w_updT + (size_t)h * (3 * HH);
#pragma unroll
    for (int kk = 0; kk < 6; ++kk)
      acc = WMMA_BF16(afr[kk], bfrag(col, kk * 32, hh), acc);
#pragma unroll
    for (int j = 0; j < 8; ++j) zfr[c][j] = sigf(acc[j]);
  }

  // ---- r gate -> rh = r*hs, transposed through per-wave LDS tile ----
  bf16* myt = &rhtile[wave][0];
#pragma unroll
  for (int c = 0; c < 4; ++c) {
    int h = c * 16 + l15;
    float bv = rst_b[h];
    f32x8 acc;
#pragma unroll
    for (int j = 0; j < 8; ++j) acc[j] = bv;
    const bf16* col = w_rstT + (size_t)h * (3 * HH);
#pragma unroll
    for (int kk = 0; kk < 6; ++kk)
      acc = WMMA_BF16(afr[kk], bfrag(col, kk * 32, hh), acc);
    const float* hsp = hsT + ((size_t)(b * HH + h)) * NN + ntile * 16 + hh * 8;
    f32x4 h0 = *(const f32x4*)(hsp);
    f32x4 h1 = *(const f32x4*)(hsp + 4);
#pragma unroll
    for (int j = 0; j < 8; ++j) {
      float hv = (j < 4) ? h0[j] : h1[j - 4];
      myt[(hh * 8 + j) * 72 + h] = (bf16)(sigf(acc[j]) * hv);
    }
  }
  __syncthreads();

  // ---- h' GEMM: A-fragments for K=128..191 come from the LDS rh tile ----
  bf16x16 r0 = afrag_bf(myt + l15 * 72, 0, hh);
  bf16x16 r1 = afrag_bf(myt + l15 * 72, 32, hh);
#pragma unroll
  for (int c = 0; c < 4; ++c) {
    int h = c * 16 + l15;
    float bv = trn_b[h];
    f32x8 acc;
#pragma unroll
    for (int j = 0; j < 8; ++j) acc[j] = bv;
    const bf16* col = w_trnT + (size_t)h * (3 * HH);
#pragma unroll
    for (int kk = 0; kk < 4; ++kk)
      acc = WMMA_BF16(afr[kk], bfrag(col, kk * 32, hh), acc);
    acc = WMMA_BF16(r0, bfrag(col, 128, hh), acc);
    acc = WMMA_BF16(r1, bfrag(col, 160, hh), acc);

    int n0 = ntile * 16 + hh * 8;
    float* hp = hsT + ((size_t)(b * HH + h)) * NN + n0;
    f32x4 o0 = *(const f32x4*)(hp), o1 = *(const f32x4*)(hp + 4);
    f32x4 n0v, n1v;
#pragma unroll
    for (int j = 0; j < 4; ++j) {
      float z0 = zfr[c][j], z1 = zfr[c][j + 4];
      n0v[j] = (1.0f - z0) * o0[j] + z0 * tanhf(acc[j]);
      n1v[j] = (1.0f - z1) * o1[j] + z1 * tanhf(acc[j + 4]);
    }
    *(f32x4*)(hp)     = n0v;
    *(f32x4*)(hp + 4) = n1v;
#pragma unroll
    for (int j = 0; j < 8; ++j) {
      float nv = (j < 4) ? n0v[j] : n1v[j - 4];
      hs_bf[((size_t)(b * NN + n0 + j)) * HH + h] = (bf16)nv;
    }
  }
}

// Final: join=[hs, ann]; out[b][h] = tanh(sum_n sigmoid(join@Wa+ba)*tanh(join@Ws+bs)).
// One block per batch, 8 waves; LDS ds_add_f32 reduction over n.
__global__ void k_final(const bf16* __restrict__ hs_bf, const bf16* __restrict__ ann_bf,
                        const bf16* __restrict__ w_attT, const float* __restrict__ att_b,
                        const bf16* __restrict__ w_stT,  const float* __restrict__ st_b,
                        float* __restrict__ out) {
  __shared__ float ssum[HH];
  int b = blockIdx.x;
  if (threadIdx.x < HH) ssum[threadIdx.x] = 0.0f;
  __syncthreads();

  int w    = threadIdx.x >> 5;
  int lane = threadIdx.x & 31;
  int l15  = lane & 15;
  int hh   = lane >> 4;

  for (int t = w; t < NN / 16; t += 8) {
    const bf16* hrow = hs_bf + ((size_t)(b * NN + t * 16 + l15)) * HH;
    const bf16* nrow = ann_bf + ((size_t)(b * NN + t * 16 + l15)) * AD;
    bf16x16 af0 = afrag_bf(hrow, 0, hh);
    bf16x16 af1 = afrag_bf(hrow, 32, hh);
    bf16x16 af2 = afrag_bf(nrow, 0, hh);
#pragma unroll
    for (int c = 0; c < 4; ++c) {
      int h = c * 16 + l15;
      float ba = att_b[h], bs = st_b[h];
      f32x8 aacc, sacc;
#pragma unroll
      for (int j = 0; j < 8; ++j) { aacc[j] = ba; sacc[j] = bs; }
      const bf16* ac = w_attT + (size_t)h * (HH + AD);
      const bf16* sc = w_stT  + (size_t)h * (HH + AD);
      aacc = WMMA_BF16(af0, bfrag(ac, 0, hh), aacc);
      aacc = WMMA_BF16(af1, bfrag(ac, 32, hh), aacc);
      aacc = WMMA_BF16(af2, bfrag(ac, 64, hh), aacc);
      sacc = WMMA_BF16(af0, bfrag(sc, 0, hh), sacc);
      sacc = WMMA_BF16(af1, bfrag(sc, 32, hh), sacc);
      sacc = WMMA_BF16(af2, bfrag(sc, 64, hh), sacc);
      float part = 0.0f;
#pragma unroll
      for (int j = 0; j < 8; ++j) part += sigf(aacc[j]) * tanhf(sacc[j]);
      atomicAdd(&ssum[h], part);   // ds_add_f32
    }
  }
  __syncthreads();
  if (threadIdx.x < HH) out[(size_t)b * HH + threadIdx.x] = tanhf(ssum[threadIdx.x]);
}

// ---------------- host launcher ----------------

extern "C" void kernel_launch(void* const* d_in, const int* in_sizes, int n_in,
                              void* d_out, int out_size, void* d_ws, size_t ws_size,
                              hipStream_t stream) {
  const float* hidden = (const float*)d_in[0];
  const float* annot  = (const float*)d_in[1];
  const float* Amat   = (const float*)d_in[2];
  const float* in_w   = (const float*)d_in[3];
  const float* in_b   = (const float*)d_in[4];
  const float* out_w  = (const float*)d_in[5];
  const float* out_b  = (const float*)d_in[6];
  const float* upd_w  = (const float*)d_in[7];
  const float* upd_b  = (const float*)d_in[8];
  const float* rst_w  = (const float*)d_in[9];
  const float* rst_b  = (const float*)d_in[10];
  const float* trn_w  = (const float*)d_in[11];
  const float* trn_b  = (const float*)d_in[12];
  const float* att_w  = (const float*)d_in[13];
  const float* att_b  = (const float*)d_in[14];
  const float* st_w   = (const float*)d_in[15];
  const float* st_b   = (const float*)d_in[16];

  char* p = (char*)d_ws;
  auto alloc = [&](size_t bytes) -> char* {
    char* r = p;
    p += (bytes + 255) & ~(size_t)255;
    return r;
  };
  float* hsT    = (float*)alloc((size_t)BB * HH * NN * 4);         // hsT[b][h][n]
  bf16*  hs_bf  = (bf16*) alloc((size_t)BB * NN * HH * 2);         // row-major
  bf16*  ann_bf = (bf16*) alloc((size_t)BB * NN * AD * 2);
  bf16*  ain_bf = (bf16*) alloc((size_t)BB * NN * 2 * HH * 2);     // [a_in|a_out]
  bf16*  incT   = (bf16*) alloc((size_t)BB * HH * EE * NN * 2);    // incT[b][h][m]
  bf16*  outgT  = (bf16*) alloc((size_t)BB * HH * EE * NN * 2);
  bf16*  w_inT  = (bf16*) alloc((size_t)EE * HH * HH * 2);
  bf16*  w_outT = (bf16*) alloc((size_t)EE * HH * HH * 2);
  bf16*  w_updT = (bf16*) alloc((size_t)HH * 3 * HH * 2);
  bf16*  w_rstT = (bf16*) alloc((size_t)HH * 3 * HH * 2);
  bf16*  w_trnT = (bf16*) alloc((size_t)HH * 3 * HH * 2);
  bf16*  w_attT = (bf16*) alloc((size_t)HH * (HH + AD) * 2);
  bf16*  w_stT  = (bf16*) alloc((size_t)HH * (HH + AD) * 2);

  // ---- prep ----
  {
    int t1 = BB * NN * HH;
    k_cvt_bf16<<<(t1 + 255) / 256, 256, 0, stream>>>(hidden, hs_bf, t1);
    int t2 = BB * NN * AD;
    k_cvt_bf16<<<(t2 + 255) / 256, 256, 0, stream>>>(annot, ann_bf, t2);
    k_tr_f32<<<(t1 + 255) / 256, 256, 0, stream>>>(hidden, hsT, BB, NN, HH);
    int tw = EE * HH * HH;
    k_tr_w<<<(tw + 255) / 256, 256, 0, stream>>>(in_w, w_inT, EE, HH, HH);
    k_tr_w<<<(tw + 255) / 256, 256, 0, stream>>>(out_w, w_outT, EE, HH, HH);
    int tg = 3 * HH * HH;
    k_tr_w<<<(tg + 255) / 256, 256, 0, stream>>>(upd_w, w_updT, 1, 3 * HH, HH);
    k_tr_w<<<(tg + 255) / 256, 256, 0, stream>>>(rst_w, w_rstT, 1, 3 * HH, HH);
    k_tr_w<<<(tg + 255) / 256, 256, 0, stream>>>(trn_w, w_trnT, 1, 3 * HH, HH);
    int ta = (HH + AD) * HH;
    k_tr_w<<<(ta + 255) / 256, 256, 0, stream>>>(att_w, w_attT, 1, HH + AD, HH);
    k_tr_w<<<(ta + 255) / 256, 256, 0, stream>>>(st_w, w_stT, 1, HH + AD, HH);
  }

  // ---- recurrent steps ----
  for (int s = 0; s < STEPS_K; ++s) {
    k_inout<<<256, 256, 0, stream>>>(hs_bf, w_inT, w_outT, in_b, out_b, incT, outgT);
    k_agg<<<128, 256, 0, stream>>>(Amat, incT, outgT, ain_bf);
    k_gru<<<64, 256, 0, stream>>>(ain_bf, hs_bf, w_updT, upd_b, w_rstT, rst_b,
                                  w_trnT, trn_b, hsT);
  }

  // ---- final attention readout ----
  k_final<<<BB, 256, 0, stream>>>(hs_bf, ann_bf, w_attT, att_b, w_stT, st_b,
                                  (float*)d_out);
}
